// TemporalAttention_87909390614789
// MI455X (gfx1250) — matrix operand order repaired and checked by reading, not compiled, not run on hardware
//
#include <hip/hip_runtime.h>

// ---------------------------------------------------------------------------
// TemporalAttention on MI455X (gfx1250, wave32, WMMA + Tensor Data Mover)
//   Q = x@Wq+bq ; K = x@Wk+bk ; V = x@Wv+bv
//   S = (Q K^T)/sqrt(H) ; P = softmax over QUERY axis (per key column)
//   out = P V
// All GEMMs: v_wmma_f32_16x16x32_bf16, tiles staged in LDS by TDM
// (tensor_load_to_lds, TENSORcnt double-buffered pipeline).
// Block tile 128x128: 8 waves, each wave = 16 rows x 128 cols (8 WMMAs/step).
// ---------------------------------------------------------------------------

#define DEVFN __device__ __forceinline__

typedef __attribute__((ext_vector_type(16))) __bf16   v16bf;
typedef __attribute__((ext_vector_type(8)))  __bf16   v8bf;
typedef __attribute__((ext_vector_type(8)))  float    v8f;
typedef __attribute__((ext_vector_type(4)))  unsigned v4u;
typedef __attribute__((ext_vector_type(8)))  int      v8i;
typedef __attribute__((ext_vector_type(4)))  int      v4i;

constexpr int BB = 8;     // batch
constexpr int TT = 2048;  // sequence
constexpr int HD = 1024;  // hidden

// ----------------------------- bf16 helpers --------------------------------
DEVFN __bf16 f2bf(float f) {
  union { float f; unsigned u; } v; v.f = f;
  unsigned r = (v.u + 0x7FFFu + ((v.u >> 16) & 1u)) >> 16;   // RNE
  union { unsigned short s; __bf16 b; } o; o.s = (unsigned short)r;
  return o.b;
}
DEVFN float bf2f(__bf16 b) {
  union { unsigned short s; __bf16 b; } i; i.b = b;
  union { unsigned u; float f; } o; o.u = ((unsigned)i.s) << 16;
  return o.f;
}

// --------------------------- WMMA fragment I/O -----------------------------
// A fragment 16x32 bf16 from an LDS tile (row-major, 32 elems/row):
//   lane L: row M=L&15; lanes<16 hold K={0..7,16..23}, lanes>=16 K={8..15,24..31}
DEVFN v16bf lds_a_frag(const __bf16* tile, int row0, int lane) {
  int m  = lane & 15;
  int kb = (lane >> 4) << 3;
  const __bf16* p = tile + (row0 + m) * 32 + kb;
  v8bf lo = *(const v8bf*)(p);
  v8bf hi = *(const v8bf*)(p + 16);
  v16bf a;
#pragma unroll
  for (int h = 0; h < 8; ++h) { a[h] = lo[h]; a[h + 8] = hi[h]; }
  return a;
}
// B fragment 32x16 bf16 from an LDS tile (row-major, 128 elems/row):
//   lane L: col N=n0+(L&15); lanes<16 hold K={0..7,16..23}, lanes>=16 K={8..15,24..31}
DEVFN v16bf lds_b_frag(const __bf16* tile, int n0, int lane) {
  int n  = n0 + (lane & 15);
  int kb = (lane >> 4) << 3;
  v16bf b;
#pragma unroll
  for (int h = 0; h < 8; ++h) {
    b[h]     = tile[(kb + h) * 128 + n];
    b[h + 8] = tile[(kb + 16 + h) * 128 + n];
  }
  return b;
}

DEVFN v8f wmma_bf16(v16bf a, v16bf b, v8f c) {
  return __builtin_amdgcn_wmma_f32_16x16x32_bf16(false, a, false, b, (short)0, c,
                                                 false, false);
}

// --------------------- Tensor Data Mover (TDM) helper ----------------------
// 2-D tile DMA: global (row-major, ld_elems leading dim, bf16) -> LDS.
// D# layout per CDNA5 ISA ch.8 (group0 128b, group1 256b, groups 2/3 zero).
// This toolchain exposes the 6-arg builtin:
//   (v4u g0, v8i g1, v4i g2, v4i g3, v8i g_extra, i32 cpol)
DEVFN void tdm_load_2d(unsigned lds_off, const __bf16* gptr,
                       int width_elems, int height_rows, int ld_elems) {
  unsigned long long ga = (unsigned long long)(size_t)gptr;
  v4u g0;
  g0[0] = 1u;                                           // count=1, user desc
  g0[1] = lds_off;                                      // lds_addr (bytes)
  g0[2] = (unsigned)(ga & 0xFFFFFFFFu);                 // global_addr[31:0]
  g0[3] = (unsigned)((ga >> 32) & 0x01FFFFFFu)          // global_addr[56:32]
        | 0x80000000u;                                  // type=2 ("image")
  unsigned td0 = (unsigned)ld_elems;                    // tensor_dim0
  unsigned td1 = 0x00100000u;                           // tensor_dim1 (huge)
  unsigned long long st0 = (unsigned long long)ld_elems;// tensor_dim0_stride
  v8i g1;
  g1[0] = 0x00010000;                                   // data_size=1 (2 bytes)
  g1[1] = (int)((td0 & 0xFFFFu) << 16);                 // [63:48]=td0.lo
  g1[2] = (int)((td0 >> 16) | ((td1 & 0xFFFFu) << 16)); // td0.hi | td1.lo
  g1[3] = (int)((td1 >> 16) | ((unsigned)width_elems << 16)); // td1.hi | tile_dim0
  g1[4] = (int)((unsigned)height_rows & 0xFFFFu);       // tile_dim1 (tile_dim2=0)
  g1[5] = (int)(st0 & 0xFFFFFFFFu);                     // stride0[31:0]
  g1[6] = (int)((st0 >> 32) & 0xFFFFu);                 // stride0[47:32]
  g1[7] = 0;
  v4i z4 = {0, 0, 0, 0};
  v8i z8 = {0, 0, 0, 0, 0, 0, 0, 0};
  __builtin_amdgcn_tensor_load_to_lds(g0, g1, z4, z4, z8, 0);
}

// ------------------------------- kernels -----------------------------------
__global__ void cvt_kernel(const float* __restrict__ src, __bf16* __restrict__ dst,
                           size_t n) {
  size_t i = (size_t)blockIdx.x * blockDim.x + threadIdx.x;
  size_t stride = (size_t)gridDim.x * blockDim.x;
  for (; i < n; i += stride) dst[i] = f2bf(src[i]);
}

// Unified tiled GEMM: C = A[M x K] * B[K x N]  (per blockIdx.z batch slice)
// Block tile 128x128, 8 waves (wave w -> rows [16w,16w+16) x 128 cols).
// Tiles staged in LDS by TDM, double-buffered over the K loop (step 32).
// mode 0: bf16 store, +bias                   (Q / V projection)
// mode 1: bf16 store transposed [H][T], +bias (K projection -> Kt)
// mode 2: bf16 store, *scale                  (scores S)
// mode 3: f32 store                           (out = P V)
__global__ void __launch_bounds__(256)
gemm_tdm_kernel(const __bf16* __restrict__ A, const __bf16* __restrict__ Bm,
                const float* __restrict__ bias, void* __restrict__ Cout,
                float scale, int K, int ldA, int ldB, int ldC,
                long long sA, long long sB, long long sC, int mode) {
  __shared__ __bf16 lds[2][128 * 32 + 32 * 128];  // A tile 8KB + B tile 8KB, x2

  const int bz   = blockIdx.z;
  const __bf16* Ab = A  + (size_t)bz * sA;
  const __bf16* Bb = Bm + (size_t)bz * sB;
  const int row0 = blockIdx.y * 128;
  const int col0 = blockIdx.x * 128;
  const int wv   = (int)(threadIdx.x >> 5);
  const int lane = (int)(threadIdx.x & 31);

  const unsigned ldsA0 = (unsigned)(size_t)(&lds[0][0]);
  const unsigned ldsB0 = ldsA0 + 128 * 32 * 2;
  const unsigned ldsA1 = (unsigned)(size_t)(&lds[1][0]);
  const unsigned ldsB1 = ldsA1 + 128 * 32 * 2;

  const int nIter = K / 32;
  if (wv == 0) {                                  // prologue: DMA tile 0 -> buf0
    tdm_load_2d(ldsA0, Ab + (size_t)row0 * ldA, 32, 128, ldA);
    tdm_load_2d(ldsB0, Bb + col0, 128, 32, ldB);
  }

  v8f c[8] = {};
  for (int it = 0; it < nIter; ++it) {
    const int buf = it & 1;
    __builtin_amdgcn_s_wait_tensorcnt(0);         // current buffer's DMA done
    __syncthreads();                              // publish LDS to all waves
    if (wv == 0 && it + 1 < nIter) {              // prefetch next tile -> other buf
      const int k = (it + 1) * 32;
      tdm_load_2d(buf ? ldsA0 : ldsA1, Ab + (size_t)row0 * ldA + k, 32, 128, ldA);
      tdm_load_2d(buf ? ldsB0 : ldsB1, Bb + (size_t)k * ldB + col0, 128, 32, ldB);
    }
    const __bf16* tA = &lds[buf][0];
    const __bf16* tB = &lds[buf][128 * 32];
    // Preload all fragments so the ds_loads issue under one wait, then the
    // 8 WMMAs chain back-to-back.
    v16bf a = lds_a_frag(tA, wv * 16, lane);
    v16bf bfr[8];
#pragma unroll
    for (int j = 0; j < 8; ++j) bfr[j] = lds_b_frag(tB, 16 * j, lane);
#pragma unroll
    for (int j = 0; j < 8; ++j) c[j] = wmma_bf16(a, bfr[j], c[j]);
    __syncthreads();                              // reads done before buf reuse
  }

  // ------------------------------ epilogue ---------------------------------
  const int n = lane & 15, mh = (lane >> 4) * 8;
  const int gr0 = row0 + wv * 16 + mh;            // global row of c[.][0]
  if (mode == 0) {
    __bf16* C = (__bf16*)Cout + (size_t)bz * sC;
#pragma unroll
    for (int j = 0; j < 8; ++j) {
      float bv = bias[col0 + 16 * j + n];
      __bf16* p = C + (size_t)gr0 * ldC + col0 + 16 * j + n;
#pragma unroll
      for (int r = 0; r < 8; ++r) p[(size_t)r * ldC] = f2bf(c[j][r] + bv);
    }
  } else if (mode == 1) {                         // Kt[b][h][t]
    __bf16* C = (__bf16*)Cout;
    int b  = gr0 / TT;
    int t0 = gr0 - b * TT;
#pragma unroll
    for (int j = 0; j < 8; ++j) {
      float bv = bias[col0 + 16 * j + n];
      __bf16* p = C + (size_t)b * HD * TT + (size_t)(col0 + 16 * j + n) * TT + t0;
#pragma unroll
      for (int r = 0; r < 8; ++r) p[r] = f2bf(c[j][r] + bv);   // contiguous in T
    }
  } else if (mode == 2) {
    __bf16* C = (__bf16*)Cout + (size_t)bz * sC;
#pragma unroll
    for (int j = 0; j < 8; ++j) {
      __bf16* p = C + (size_t)gr0 * ldC + col0 + 16 * j + n;
#pragma unroll
      for (int r = 0; r < 8; ++r) p[(size_t)r * ldC] = f2bf(c[j][r] * scale);
    }
  } else {
    float* C = (float*)Cout + (size_t)bz * sC;
#pragma unroll
    for (int j = 0; j < 8; ++j) {
      float* p = C + (size_t)gr0 * ldC + col0 + 16 * j + n;
#pragma unroll
      for (int r = 0; r < 8; ++r) p[(size_t)r * ldC] = c[j][r];
    }
  }
}

// Softmax over the QUERY axis: for each (b, k) column of S[b, :, k].
// One thread per column; consecutive threads -> consecutive k -> coalesced.
__global__ void colsoftmax_kernel(__bf16* __restrict__ S) {
  int idx = (int)(blockIdx.x * blockDim.x + threadIdx.x);  // 0 .. B*T-1
  int b = idx / TT, k = idx % TT;
  __bf16* col = S + (size_t)b * TT * TT + k;

  float m = -3.402823466e38f;
  for (int q = 0; q < TT; ++q) m = fmaxf(m, bf2f(col[(size_t)q * TT]));
  float z = 0.0f;
  for (int q = 0; q < TT; ++q) z += __expf(bf2f(col[(size_t)q * TT]) - m);
  float rz = 1.0f / z;
  for (int q = 0; q < TT; ++q) {
    size_t o = (size_t)q * TT;
    col[o] = f2bf(__expf(bf2f(col[o]) - m) * rz);
  }
}

// ------------------------------- launcher ----------------------------------
extern "C" void kernel_launch(void* const* d_in, const int* in_sizes, int n_in,
                              void* d_out, int out_size, void* d_ws, size_t ws_size,
                              hipStream_t stream) {
  const float* x  = (const float*)d_in[0];
  const float* Wq = (const float*)d_in[1];
  const float* bq = (const float*)d_in[2];
  const float* Wk = (const float*)d_in[3];
  const float* bk = (const float*)d_in[4];
  const float* Wv = (const float*)d_in[5];
  const float* bv = (const float*)d_in[6];
  float* out = (float*)d_out;

  // Workspace (bf16 elems): xb | Wqb | Wkb | Wvb | Qb | Ktb | Vb | Sb (~208 MB)
  const size_t NX = (size_t)BB * TT * HD;
  const size_t NW = (size_t)HD * HD;
  __bf16* xb  = (__bf16*)d_ws;
  __bf16* wqb = xb  + NX;
  __bf16* wkb = wqb + NW;
  __bf16* wvb = wkb + NW;
  __bf16* qb  = wvb + NW;
  __bf16* ktb = qb  + NX;
  __bf16* vb  = ktb + NX;
  __bf16* sb  = vb  + NX;

  // 1) downconvert inputs to bf16
  cvt_kernel<<<4096, 256, 0, stream>>>(x,  xb,  NX);
  cvt_kernel<<<512,  256, 0, stream>>>(Wq, wqb, NW);
  cvt_kernel<<<512,  256, 0, stream>>>(Wk, wkb, NW);
  cvt_kernel<<<512,  256, 0, stream>>>(Wv, wvb, NW);

  // 2) projections: C[16384 x 1024] = xb @ W + b
  gemm_tdm_kernel<<<dim3(8, 128, 1), 256, 0, stream>>>(
      xb, wqb, bq, qb, 1.0f, HD, HD, HD, HD, 0, 0, 0, 0);
  gemm_tdm_kernel<<<dim3(8, 128, 1), 256, 0, stream>>>(
      xb, wkb, bk, ktb, 1.0f, HD, HD, HD, HD, 0, 0, 0, 1);   // K stored [H][T]
  gemm_tdm_kernel<<<dim3(8, 128, 1), 256, 0, stream>>>(
      xb, wvb, bv, vb, 1.0f, HD, HD, HD, HD, 0, 0, 0, 0);

  // 3) scores: per batch, S[T x T] = (Q @ Kt) * 1/sqrt(H)
  gemm_tdm_kernel<<<dim3(16, 16, BB), 256, 0, stream>>>(
      qb, ktb, nullptr, sb, 0.03125f, HD, HD, TT, TT,
      (long long)TT * HD, (long long)HD * TT, (long long)TT * TT, 2);

  // 4) per-key-column softmax over queries (axis=1), in place
  colsoftmax_kernel<<<(BB * TT) / 256, 256, 0, stream>>>(sb);

  // 5) out = P @ V : per batch, [T x H], f32
  gemm_tdm_kernel<<<dim3(8, 16, BB), 256, 0, stream>>>(
      sb, vb, nullptr, out, 1.0f, TT, TT, HD, HD,
      (long long)TT * TT, (long long)TT * HD, (long long)TT * HD, 3);
}
